// Res_Module_55688545960610
// MI455X (gfx1250) — matrix operand order repaired and checked by reading, not compiled, not run on hardware
//
#include <hip/hip_runtime.h>

// Problem constants (match reference)
#define NB 16
#define N_PER 2048
#define KNN 16
#define D_IN 64
#define D_H 128
#define D_OUT 256

typedef __attribute__((ext_vector_type(16))) __bf16 v16bf;
typedef __attribute__((ext_vector_type(8)))  __bf16 v8bf;
typedef __attribute__((ext_vector_type(2)))  __bf16 v2bf;
typedef __attribute__((ext_vector_type(8)))  float  v8f;

// ---------------------------------------------------------------------------
// Kernel 1: per-graph KNN (K=16 nearest incl. self). One block = 256 targets.
// ---------------------------------------------------------------------------
__global__ __launch_bounds__(256)
void knn_kernel(const float* __restrict__ pos, int* __restrict__ idx) {
  __shared__ float ppos[N_PER * 3];
  const int chunks = N_PER / 256;
  const int b = blockIdx.x / chunks;
  const int chunk = blockIdx.x % chunks;
  const float* gp = pos + (size_t)b * N_PER * 3;
  for (int i = threadIdx.x; i < N_PER * 3; i += 256) ppos[i] = gp[i];
  __syncthreads();

  const int ti = chunk * 256 + threadIdx.x;
  const float px = ppos[ti * 3 + 0];
  const float py = ppos[ti * 3 + 1];
  const float pz = ppos[ti * 3 + 2];

  float bd[KNN];
  int   bi[KNN];
#pragma unroll
  for (int s = 0; s < KNN; ++s) { bd[s] = 3.0e38f; bi[s] = 0; }

  for (int j = 0; j < N_PER; ++j) {
    const float dx = ppos[j * 3 + 0] - px;
    const float dy = ppos[j * 3 + 1] - py;
    const float dz = ppos[j * 3 + 2] - pz;
    const float d2 = dx * dx + dy * dy + dz * dz;
    if (d2 < bd[KNN - 1]) {
      bd[KNN - 1] = d2; bi[KNN - 1] = j;
#pragma unroll
      for (int s = KNN - 1; s > 0; --s) {
        if (bd[s] < bd[s - 1]) {
          float td = bd[s]; bd[s] = bd[s - 1]; bd[s - 1] = td;
          int   tt = bi[s]; bi[s] = bi[s - 1]; bi[s - 1] = tt;
        }
      }
    }
  }
  int* op = idx + (size_t)(b * N_PER + ti) * KNN;
#pragma unroll
  for (int s = 0; s < KNN; ++s) op[s] = bi[s];
}

// ---------------------------------------------------------------------------
// Kernel 2: fused edge-feature build + MLP(67->128->128) + max-agg + 128->256.
// One block (8 wave32) per TI=16 targets -> 256 edge rows.
// Weights are staged in LDS PRE-PACKED in WMMA B-fragment order
// ([kchunk][ntile][lane][16 elems]) so each lane loads its whole fragment
// with two ds_load_b128 and no cross-half merges.
// ---------------------------------------------------------------------------
#define TI 16               // targets per block
#define ME (TI * KNN)       // 256 edge rows
#define KP1 96              // layer-1 K dim padded 67 -> 96 (3 chunks of 32)

// helper: build v16bf fragment from two contiguous v8bf LDS loads
__device__ __forceinline__ v16bf frag16(const v8bf* row, int i0, int i1) {
  const v8bf a0 = row[i0];
  const v8bf a1 = row[i1];
  return __builtin_shufflevector(a0, a1, 0, 1, 2, 3, 4, 5, 6, 7,
                                 8, 9, 10, 11, 12, 13, 14, 15);
}

__global__ __launch_bounds__(256)
void pointconv_fused(const float* __restrict__ x, const float* __restrict__ pos,
                     const int* __restrict__ idx,
                     const float* __restrict__ W1, const float* __restrict__ b1,
                     const float* __restrict__ W2, const float* __restrict__ b2,
                     const float* __restrict__ Wg, const float* __restrict__ bg,
                     float* __restrict__ out) {
  // Weights pre-packed as B-fragments: [kc][nt][lane(32)][e(16)]
  __shared__ v16bf sW1p[3 * 8 * 32];    // 24 KB
  __shared__ v16bf sW2p[4 * 8 * 32];    // 32 KB
  __shared__ v16bf sWgp[4 * 16 * 32];   // 64 KB
  __shared__ __bf16 sA[ME * KP1];       // 48 KB  edge features (row-major)
  __shared__ __bf16 sH[ME * D_H];       // 64 KB  hidden activations
  __shared__ __bf16 sAgg[TI * D_H];     //  4 KB  max-aggregated

  const int tid  = threadIdx.x;
  const int lane = tid & 31;
  const int wave = tid >> 5;            // 0..7
  const int nlo  = lane & 15;
  const int khalf = lane >> 4;          // 0 or 1

  const int blk = blockIdx.x;           // 0 .. NB*N_PER/TI - 1
  const int b   = blk / (N_PER / TI);
  const int t0  = (blk % (N_PER / TI)) * TI;

  // ---- Stage weights, packing into B-fragment order (coalesced reads) ----
  {
    __bf16* p1 = (__bf16*)sW1p;
    for (int i = tid; i < KP1 * D_H; i += 256) {       // virtual 96x128
      const int k = i >> 7, n = i & 127;
      const float v = (k < D_IN + 3) ? W1[k * D_H + n] : 0.0f;
      const int kc = k >> 5, khf = (k >> 4) & 1, e = k & 15;
      const int nt = n >> 4, nl = n & 15;
      p1[(((kc * 8 + nt) * 32) + (khf * 16 + nl)) * 16 + e] = (__bf16)v;
    }
    __bf16* p2 = (__bf16*)sW2p;
    for (int i = tid; i < D_H * D_H; i += 256) {       // 128x128
      const int k = i >> 7, n = i & 127;
      const int kc = k >> 5, khf = (k >> 4) & 1, e = k & 15;
      const int nt = n >> 4, nl = n & 15;
      p2[(((kc * 8 + nt) * 32) + (khf * 16 + nl)) * 16 + e] = (__bf16)W2[i];
    }
    __bf16* pg = (__bf16*)sWgp;
    for (int i = tid; i < D_H * D_OUT; i += 256) {     // 128x256
      const int k = i >> 8, n = i & 255;
      const int kc = k >> 5, khf = (k >> 4) & 1, e = k & 15;
      const int nt = n >> 4, nl = n & 15;
      pg[(((kc * 16 + nt) * 32) + (khf * 16 + nl)) * 16 + e] = (__bf16)Wg[i];
    }
  }

  // ---- Build edge feature rows: [x_j(64) | pos_j - pos_i (3) | 0 pad] ----
  {
    const int e = tid;                 // one of 256 edge rows
    const int t = e >> 4, k = e & 15;
    const int tg = b * N_PER + t0 + t;
    const int src = idx[(size_t)tg * KNN + k];
    const int sg = b * N_PER + src;
    const float rx = pos[sg * 3 + 0] - pos[tg * 3 + 0];
    const float ry = pos[sg * 3 + 1] - pos[tg * 3 + 1];
    const float rz = pos[sg * 3 + 2] - pos[tg * 3 + 2];
    const float* xr = x + (size_t)sg * D_IN;
    v2bf* rowp = (v2bf*)(sA + (size_t)e * KP1);        // 4B-aligned (192B rows)
#pragma unroll
    for (int c = 0; c < D_IN; c += 2) {
      v2bf p; p.x = (__bf16)xr[c]; p.y = (__bf16)xr[c + 1];
      rowp[c >> 1] = p;
    }
    { v2bf p; p.x = (__bf16)rx; p.y = (__bf16)ry; rowp[32] = p; }
    { v2bf p; p.x = (__bf16)rz; p.y = (__bf16)0.0f; rowp[33] = p; }
    const v2bf z = {(__bf16)0.0f, (__bf16)0.0f};
#pragma unroll
    for (int c = 68; c < KP1; c += 2) rowp[c >> 1] = z;
  }
  __syncthreads();

  // -------- Layer 1: H = relu(A @ W1 + b1)   M=256 N=128 K=96 -------------
  for (int mi = 0; mi < 2; ++mi) {
    const int mt = wave * 2 + mi;
    const int m = mt * 16 + nlo;                       // A-frag row
    const v8bf* arow = (const v8bf*)(sA + (size_t)m * KP1);  // 12 chunks of 8
    for (int nt = 0; nt < 8; ++nt) {
      v8f acc = {};
#pragma unroll
      for (int kc = 0; kc < 3; ++kc) {
        const v16bf a = frag16(arow, kc * 4 + 2 * khalf, kc * 4 + 2 * khalf + 1);
        const v16bf bm = sW1p[(kc * 8 + nt) * 32 + lane];
        acc = __builtin_amdgcn_wmma_f32_16x16x32_bf16(
            false, a, false, bm, (short)0, acc, false, false);
      }
      const int gn = nt * 16 + nlo;
      const float bias = b1[gn];
#pragma unroll
      for (int r = 0; r < 8; ++r) {
        float v = acc[r] + bias;
        v = v > 0.f ? v : 0.f;
        const int row = mt * 16 + r + 8 * khalf;       // C layout: m=r+8*khalf
        sH[row * D_H + gn] = (__bf16)v;
      }
    }
  }
  __syncthreads();

  // -------- Layer 2 + max over 16 neighbors (one M-tile = one target) -----
  for (int mi = 0; mi < 2; ++mi) {
    const int mt = wave * 2 + mi;                      // local target index
    const int m = mt * 16 + nlo;
    const v8bf* hrow = (const v8bf*)(sH + (size_t)m * D_H);  // 16 chunks of 8
    for (int nt = 0; nt < 8; ++nt) {
      v8f acc = {};
#pragma unroll
      for (int kc = 0; kc < 4; ++kc) {
        const v16bf a = frag16(hrow, kc * 4 + 2 * khalf, kc * 4 + 2 * khalf + 1);
        const v16bf bm = sW2p[(kc * 8 + nt) * 32 + lane];
        acc = __builtin_amdgcn_wmma_f32_16x16x32_bf16(
            false, a, false, bm, (short)0, acc, false, false);
      }
      const int gn = nt * 16 + nlo;
      const float bias = b2[gn];
      float cmax = -3.0e38f;
#pragma unroll
      for (int r = 0; r < 8; ++r) {
        float v = acc[r] + bias;
        v = v > 0.f ? v : 0.f;
        cmax = fmaxf(cmax, v);
      }
      // lanes l and l^16 hold rows m and m+8 of the same column: combine
      const float other = __shfl_xor(cmax, 16, 32);
      cmax = fmaxf(cmax, other);
      if (khalf == 0) sAgg[mt * D_H + gn] = (__bf16)cmax;
    }
  }
  __syncthreads();

  // -------- Layer 3: out = relu(Agg @ Wg + bg)  M=16 N=256 K=128 ----------
  {
    const int m = nlo;                                 // Agg row (16 targets)
    const v8bf* grow = (const v8bf*)(sAgg + (size_t)m * D_H);
    for (int ni = 0; ni < 2; ++ni) {
      const int nt = wave * 2 + ni;                    // 16 N-tiles = 256 cols
      const int gn = nt * 16 + nlo;
      v8f acc = {};
#pragma unroll
      for (int kc = 0; kc < 4; ++kc) {
        const v16bf a = frag16(grow, kc * 4 + 2 * khalf, kc * 4 + 2 * khalf + 1);
        const v16bf bm = sWgp[(kc * 16 + nt) * 32 + lane];
        acc = __builtin_amdgcn_wmma_f32_16x16x32_bf16(
            false, a, false, bm, (short)0, acc, false, false);
      }
      const float bias = bg[gn];
#pragma unroll
      for (int r = 0; r < 8; ++r) {
        float v = acc[r] + bias;
        v = v > 0.f ? v : 0.f;
        const int row = r + 8 * khalf;                 // target within tile
        out[(size_t)(b * N_PER + t0 + row) * D_OUT + gn] = v;
      }
    }
  }
}

// ---------------------------------------------------------------------------
// Kernel 3: copy pos (float) and batch (int bit patterns) to the tuple tail.
// ---------------------------------------------------------------------------
__global__ __launch_bounds__(256)
void tail_copy(const float* __restrict__ pos, const int* __restrict__ batch,
               float* __restrict__ out_pos, int* __restrict__ out_batch) {
  const int i = blockIdx.x * blockDim.x + threadIdx.x;
  const int NT = NB * N_PER;
  if (i < NT * 3) out_pos[i] = pos[i];
  if (i < NT) out_batch[i] = batch[i];
}

extern "C" void kernel_launch(void* const* d_in, const int* in_sizes, int n_in,
                              void* d_out, int out_size, void* d_ws, size_t ws_size,
                              hipStream_t stream) {
  const float* x   = (const float*)d_in[0];
  const float* pos = (const float*)d_in[1];
  const int*   bat = (const int*)d_in[2];
  const float* W1  = (const float*)d_in[3];
  const float* b1  = (const float*)d_in[4];
  const float* W2  = (const float*)d_in[5];
  const float* b2  = (const float*)d_in[6];
  const float* Wg  = (const float*)d_in[7];
  const float* bg  = (const float*)d_in[8];
  float* out = (float*)d_out;

  int* idx = (int*)d_ws;   // NB*N_PER*KNN ints = 2 MB scratch

  knn_kernel<<<NB * (N_PER / 256), 256, 0, stream>>>(pos, idx);

  pointconv_fused<<<NB * (N_PER / TI), 256, 0, stream>>>(
      x, pos, idx, W1, b1, W2, b2, Wg, bg, out);

  const int NT = NB * N_PER;
  float* out_pos = out + (size_t)NT * D_OUT;
  int*   out_bat = (int*)(out + (size_t)NT * (D_OUT + 3));
  tail_copy<<<(NT * 3 + 255) / 256, 256, 0, stream>>>(pos, bat, out_pos, out_bat);
}
